// PreCorrectorMLP_StaticDiag_79113297592679
// MI455X (gfx1250) — compile-verified, matrix operand order checked by reading, not run on hardware
//
#include <hip/hip_runtime.h>

typedef __attribute__((ext_vector_type(16))) _Float16 v16h;
typedef __attribute__((ext_vector_type(8)))  float    v8f;
typedef __attribute__((ext_vector_type(8)))  unsigned v8u;

#define WAVE 32

// ---------------------------------------------------------------- ws init ---
__global__ void zero_ws_kernel(unsigned* ws) {
    if (blockIdx.x == 0 && threadIdx.x == 0) ws[0] = 0u;
}

// ---------------------------------------------------- abs-max reduction -----
// norm = max(|edges[n0 .. n0+n)|) ; stored as float bits via atomic u32 max
// (valid ordering because all values are non-negative after fabsf).
__global__ void absmax_kernel(const float* __restrict__ e, int n0, int n,
                              unsigned* __restrict__ ws) {
    const int stride = gridDim.x * blockDim.x;
    const int i = blockIdx.x * blockDim.x + threadIdx.x;
    float m = 0.0f;
    const float* base = e + n0;
    if ((((uintptr_t)base) & 15u) == 0) {
        const int nv = n >> 2;
        const float4* p = (const float4*)base;
        for (int t = i; t < nv; t += stride) {
            float4 v = p[t];
            m = fmaxf(m, fmaxf(fmaxf(fabsf(v.x), fabsf(v.y)),
                               fmaxf(fabsf(v.z), fabsf(v.w))));
        }
        for (int t = (nv << 2) + i; t < n; t += stride)
            m = fmaxf(m, fabsf(base[t]));
    } else {
        for (int t = i; t < n; t += stride)
            m = fmaxf(m, fabsf(base[t]));
    }
    #pragma unroll
    for (int off = 16; off > 0; off >>= 1)
        m = fmaxf(m, __shfl_xor(m, off, WAVE));
    if ((threadIdx.x & (WAVE - 1)) == 0)
        atomicMax(ws, __float_as_uint(m));
}

// ------------------------------------------------------------- head copy ----
__global__ void copy_head_kernel(const float* __restrict__ in,
                                 float* __restrict__ out, int n) {
    const int i = blockIdx.x * blockDim.x + threadIdx.x;
    const int nv = n >> 2;
    if (i < nv) ((float4*)out)[i] = ((const float4*)in)[i];
    const int rem = n - (nv << 2);
    if (i < rem) out[(nv << 2) + i] = in[(nv << 2) + i];
}

// ----------------------------------------------- tail: MLP via WMMA ---------
// Per lane: one tail element. Two v_wmma_f32_16x16x32_f16 per wave cover all
// 32 elements (A = W2 in f16, B = per-element h1 vectors, C = 0, f32 accum).
__global__ void mlp_tail_kernel(const float* __restrict__ edges,
                                float* __restrict__ out,
                                const float* __restrict__ W1,
                                const float* __restrict__ b1,
                                const float* __restrict__ W2,
                                const float* __restrict__ b2,
                                const float* __restrict__ W3,
                                const float* __restrict__ b3,
                                const float* __restrict__ alpha_p,
                                const unsigned* __restrict__ ws,
                                int n0, int n) {
    const int lane = threadIdx.x & (WAVE - 1);
    const int tid  = blockIdx.x * blockDim.x + threadIdx.x;
    const int wbase = tid & ~(WAVE - 1);     // first tail-element of this wave
    if (wbase >= n) return;                  // uniform per wave: EXEC stays full

    const float norm  = __uint_as_float(ws[0]);
    const float inv   = 1.0f / norm;
    const float scale = alpha_p[0] * norm;

    // --- load this lane's edge value (lane-masked; reconverges before WMMA)
    const int t = tid;
    const float e = (t < n) ? edges[n0 + t] : 0.0f;
    const float x = e * inv;

    // --- layer 1 in f32, pack h1[0..7] into 4 dwords of f16 pairs
    unsigned hp[4];
    #pragma unroll
    for (int k = 0; k < 4; ++k) {
        float f0 = fmaxf(W1[2 * k]     * x + b1[2 * k],     0.0f);
        float f1 = fmaxf(W1[2 * k + 1] * x + b1[2 * k + 1], 0.0f);
        _Float16 h0 = (_Float16)f0;
        _Float16 h1 = (_Float16)f1;
        hp[k] = (unsigned)__builtin_bit_cast(unsigned short, h0)
              | ((unsigned)__builtin_bit_cast(unsigned short, h1) << 16);
    }

    // --- A = W2 (16x32 f16, only M<8, K<8 nonzero)
    // A layout: lanes 0-15 are rows M=0..15; VGPR v holds K=2v,2v+1 (v<4);
    // VGPRs 4-7 (K=16..31) and lanes 16-31 (K=8..15 / 24..31) are zero.
    v8u au = {0, 0, 0, 0, 0, 0, 0, 0};
    if (lane < 8) {
        #pragma unroll
        for (int k = 0; k < 4; ++k) {
            _Float16 w0 = (_Float16)W2[lane * 8 + 2 * k];
            _Float16 w1 = (_Float16)W2[lane * 8 + 2 * k + 1];
            au[k] = (unsigned)__builtin_bit_cast(unsigned short, w0)
                  | ((unsigned)__builtin_bit_cast(unsigned short, w1) << 16);
        }
    }
    const v16h A = __builtin_bit_cast(v16h, au);

    // --- B matrices (32x16 f16): lanes 0-15 carry K=0..7 (h1[0..7]) for column N=lane.
    // B0 <- elements wbase+0..15 (this lane's own h1), B1 <- wbase+16..31
    // (shuffled down from the upper half so every lane did useful layer-1 work).
    v8u b0u = {0, 0, 0, 0, 0, 0, 0, 0};
    v8u b1u = {0, 0, 0, 0, 0, 0, 0, 0};
    #pragma unroll
    for (int k = 0; k < 4; ++k) {
        unsigned up = (unsigned)__shfl((int)hp[k], lane + 16, WAVE);
        b0u[k] = (lane < 16) ? hp[k] : 0u;
        b1u[k] = (lane < 16) ? up    : 0u;
    }
    const v16h B0 = __builtin_bit_cast(v16h, b0u);
    const v16h B1 = __builtin_bit_cast(v16h, b1u);

    const v8f c0 = {0.f, 0.f, 0.f, 0.f, 0.f, 0.f, 0.f, 0.f};
    // D = A x B + 0, f32 accumulate (EXEC all ones here)
    v8f d0 = __builtin_amdgcn_wmma_f32_16x16x32_f16(false, A, false, B0,
                                                    (short)0, c0, false, false);
    v8f d1 = __builtin_amdgcn_wmma_f32_16x16x32_f16(false, A, false, B1,
                                                    (short)0, c0, false, false);

    // --- layer 3: y = W3 . relu(D + b2) + b3
    // C/D layout: VGPR c = row M=c (c<8) for lanes 0-15, column N=lane.
    float y0 = b3[0], y1 = b3[0];
    #pragma unroll
    for (int c = 0; c < 8; ++c) {
        const float w3c = W3[c];
        const float b2c = b2[c];
        y0 += w3c * fmaxf(d0[c] + b2c, 0.0f);
        y1 += w3c * fmaxf(d1[c] + b2c, 0.0f);
    }
    // route upper-half results back so the store is one coalesced b32/lane
    const float yup = __shfl(y1, lane & 15, WAVE);
    const float y   = (lane < 16) ? y0 : yup;

    if (t < n) out[n0 + t] = e + scale * y;
}

// ---------------------------------------------------------------------------
extern "C" void kernel_launch(void* const* d_in, const int* in_sizes, int n_in,
                              void* d_out, int out_size, void* d_ws, size_t ws_size,
                              hipStream_t stream) {
    const float* edges = (const float*)d_in[1];
    const float* W1    = (const float*)d_in[4];
    const float* b1    = (const float*)d_in[5];
    const float* W2    = (const float*)d_in[6];
    const float* b2    = (const float*)d_in[7];
    const float* W3    = (const float*)d_in[8];
    const float* b3    = (const float*)d_in[9];
    const float* alpha = (const float*)d_in[10];

    const int n_nodes = in_sizes[0];
    const int n_edges = in_sizes[1];
    const int n_tail  = n_edges - n_nodes;

    float*    out = (float*)d_out;
    unsigned* ws  = (unsigned*)d_ws;

    // 1) zero the norm accumulator (re-done every call: deterministic)
    hipLaunchKernelGGL(zero_ws_kernel, dim3(1), dim3(1), 0, stream, ws);

    // 2) abs-max over the off-diagonal tail
    hipLaunchKernelGGL(absmax_kernel, dim3(1024), dim3(256), 0, stream,
                       edges, n_nodes, n_tail, ws);

    // 3) copy diagonal head unchanged
    {
        int nv = n_nodes >> 2;
        int blocks = (nv + 255) / 256;
        if (blocks < 1) blocks = 1;
        hipLaunchKernelGGL(copy_head_kernel, dim3(blocks), dim3(256), 0, stream,
                           edges, out, n_nodes);
    }

    // 4) MLP correction on the tail (WMMA path)
    {
        long long threads = ((long long)n_tail + WAVE - 1) & ~(long long)(WAVE - 1);
        int blocks = (int)((threads + 255) / 256);
        if (blocks < 1) blocks = 1;
        hipLaunchKernelGGL(mlp_tail_kernel, dim3(blocks), dim3(256), 0, stream,
                           edges, out, W1, b1, W2, b2, W3, b3, alpha, ws,
                           n_nodes, n_tail);
    }
}
